// SimpleFriendshipGraph_52158082842897
// MI455X (gfx1250) — matrix operand order repaired and checked by reading, not compiled, not run on hardware
//
#include <hip/hip_runtime.h>
#include <hip/hip_bf16.h>

// ---------------------------------------------------------------------------
// SimpleFriendshipGraph for MI455X (gfx1250, wave32, WMMA)
//
// out = (F[user_ids] @ E) @ W^T + b
//   B=2048, N=10100 (K of main GEMM), D=128
//
// Roofline: gathered-F traffic = 2048*10100*4B = 83 MB -> 3.6us @ 23.3 TB/s.
// fp32 math would be compute bound (~tens of us); bf16 WMMA (16x16x32, f32
// accum) pushes compute under the memory floor -> HBM-bound kernel.
//
// Loop schedule (branch-free steady state, last K-step peeled):
//   issue next-A global load        (staging, fp32)
//   issue current-B global loads    (bf16, 2x16B contiguous)
//   ds_load_b128 x4 current A fragments
//   v_wmma_f32_16x16x32_bf16 x2     (its loadcnt wait covers the staging
//                                    load too, since loads retire in order)
//   v_cvt_pk_bf16_f32 + ds_store    (no extra wait by then)
//   barrier, flip buffer
// ---------------------------------------------------------------------------

typedef __attribute__((ext_vector_type(16))) __bf16   v16bf;
typedef __attribute__((ext_vector_type(8)))  __bf16   bf16x8;
typedef __attribute__((ext_vector_type(2)))  __bf16   bf16x2;
typedef __attribute__((ext_vector_type(8)))  float    v8f;
typedef __attribute__((ext_vector_type(4)))  unsigned u32x4;

#define N_TOT   10100          // graph nodes (K of the main GEMM)
#define KPAD    10112          // N_TOT rounded up to 32 (one WMMA K-step)
#define DIM     128            // embedding dim
#define BATCH   2048           // user ids
#define KSTEPS  (KPAD / 32)    // 316 K-steps of 32
#define KSPLIT  8              // split-K factor (fills more WGPs)
#define SPLIT_STEPS 40         // ceil(316/8); last split does 36

// workspace layout (bytes, 16B aligned)
#define EBF_BYTES   ((size_t)DIM * KPAD * 2)          // 2,588,672
#define WBF_BYTES   ((size_t)DIM * DIM * 2)           //    32,768
#define PARTS_BYTES ((size_t)KSPLIT * BATCH * DIM * 4)// 8,388,608

static __device__ __forceinline__ unsigned pack2(float a, float b) {
  bf16x2 t; t[0] = (__bf16)a; t[1] = (__bf16)b;
  return __builtin_bit_cast(unsigned, t);
}
static __device__ __forceinline__ v16bf cat8(bf16x8 lo, bf16x8 hi) {
  return __builtin_shufflevector(lo, hi, 0,1,2,3,4,5,6,7,8,9,10,11,12,13,14,15);
}
// A-fragment from LDS. ISA 7.12.2, 16-bit A 16x32: lane(hi=0) VGPR0..3 hold
// K={0..7} pairs, VGPR4..7 K={16..23}; lane(hi=1) K={8..15},{24..31}.
// LDS holds the tile row-major bf16[32][32] packed as u32 pairs, so a lane's
// 8 VGPRs are exactly two contiguous ds_load_b128 at u32 idx row*16+hi*4 and +8.
static __device__ __forceinline__ v16bf lds_loadA(const unsigned* p) {
  u32x4 lo = *(const u32x4*)p;
  u32x4 hi = *(const u32x4*)(p + 8);
  return cat8(__builtin_bit_cast(bf16x8, lo), __builtin_bit_cast(bf16x8, hi));
}
// B-fragment: lane (col n = lane&15, hi = lane>>4) holds K = hi*8 + 0..7 and
// 16 + hi*8 + 0..7 (same K order as the A fragment). With B stored K-contiguous
// per column, that is two 16B global loads at +0 and +16 bf16.
static __device__ __forceinline__ v16bf g_loadB(const __bf16* p) {
  bf16x8 lo = *(const bf16x8*)p;
  bf16x8 hi = *(const bf16x8*)(p + 16);
  return cat8(lo, hi);
}

// ---- prep: E (N_TOT x 128 f32) -> Ebf_t (128 x KPAD bf16), zero padded ----
__global__ __launch_bounds__(256) void fg_prepE(const float* __restrict__ E,
                                                __bf16* __restrict__ ebf) {
  __shared__ float t[16][17];
  const int tid = threadIdx.x;
  const int k0 = blockIdx.x * 16, d0 = blockIdx.y * 16;
  const int r = tid >> 4, c = tid & 15;
  const int k = k0 + r;
  t[r][c] = (k < N_TOT) ? E[(size_t)k * DIM + d0 + c] : 0.0f;
  __syncthreads();
  ebf[(size_t)(d0 + r) * KPAD + k0 + c] = (__bf16)t[c][r];
}

// ---- prep: W (128x128 f32) -> bf16 copy (no transpose: B(k,n)=W[n][k]) ----
__global__ __launch_bounds__(256) void fg_prepW(const float* __restrict__ W,
                                                __bf16* __restrict__ wbf) {
  const int i = blockIdx.x * 256 + threadIdx.x;
  if (i < DIM * DIM) wbf[i] = (__bf16)W[i];
}

// ---- main GEMM: parts[ks] = gather(F)[32-row block] @ Ebf_t^T (K slice) ----
// grid (64 row-blocks, KSPLIT), 256 threads = 8 waves. Wave w owns output
// columns 16w..16w+15 and two 16x16 accumulators (rows 0..15 / 16..31).
__global__ __launch_bounds__(256) void fg_main(const int* __restrict__ ids,
                                               const float* __restrict__ F,
                                               const __bf16* __restrict__ ebf,
                                               float* __restrict__ parts) {
  __shared__ unsigned ldsA[2][32 * 16]; // double-buffered 32x32 bf16 A tile
  __shared__ int sids[32];
  const int tid    = threadIdx.x;
  const int rowblk = blockIdx.x;   // 0..63
  const int ks     = blockIdx.y;   // 0..KSPLIT-1

  if (tid < 32) sids[tid] = ids[rowblk * 32 + tid];
  __syncthreads();

  // loader role: thread t fills row t>>3, float4 at col 4*(t&7) each K-step
  const int    lrow = tid >> 3;
  const int    q    = (tid & 7) * 4;
  const float* frow = F + (size_t)sids[lrow] * N_TOT;
  unsigned* slot0 = &ldsA[0][lrow * 16 + (q >> 1)];
  unsigned* slot1 = &ldsA[1][lrow * 16 + (q >> 1)];

  const int s0 = ks * SPLIT_STEPS;
  const int s1 = min(KSTEPS, s0 + SPLIT_STEPS);

  // compute role
  const int lane = tid & 31, wave = tid >> 5;
  const int ln = lane & 15, hi = lane >> 4;
  const int col0 = wave * 16;
  const __bf16* bp = ebf + (size_t)(col0 + ln) * KPAD + (size_t)hi * 8;

  v8f acc0 = {0.f,0.f,0.f,0.f,0.f,0.f,0.f,0.f};
  v8f acc1 = {0.f,0.f,0.f,0.f,0.f,0.f,0.f,0.f};

  // prologue: stage A tile s0 into buffer 0 (one-time exposed latency).
  // K tail past N_TOT reads valid in-buffer floats of later F rows; the
  // products vanish because the Ebf_t pad region is zero.
  {
    float4 v = *(const float4*)(frow + (size_t)s0 * 32 + q);
    slot0[0] = pack2(v.x, v.y);
    slot0[1] = pack2(v.z, v.w);
  }
  __syncthreads();

  int buf = 0;
  for (int s = s0; s + 1 < s1; ++s) {       // branch-free steady state
    // 1) issue next A-tile load first; B loads issue right behind it, so
    //    the WMMA's in-order loadcnt wait covers both (latencies overlap).
    float4 vnext = *(const float4*)(frow + (size_t)(s + 1) * 32 + q);
    __builtin_prefetch(frow + (size_t)(s + 2) * 32 + q, 0, 1); // global_prefetch_b8
    v16bf b = g_loadB(bp + (size_t)s * 32);

    // 2) compute on current LDS buffer
    const unsigned* ab = &ldsA[buf][0];
    v16bf a0 = lds_loadA(ab + (ln     ) * 16 + hi * 4);
    v16bf a1 = lds_loadA(ab + (ln + 16) * 16 + hi * 4);
    acc0 = __builtin_amdgcn_wmma_f32_16x16x32_bf16(false, a0, false, b,
                                                   (short)0, acc0, false, false);
    acc1 = __builtin_amdgcn_wmma_f32_16x16x32_bf16(false, a1, false, b,
                                                   (short)0, acc1, false, false);

    // 3) convert + stage next A tile (its load landed during the WMMA wait)
    unsigned* d = buf ? slot0 : slot1;
    d[0] = pack2(vnext.x, vnext.y);
    d[1] = pack2(vnext.z, vnext.w);
    __syncthreads();
    buf ^= 1;
  }
  { // peeled final K-step: no staging
    v16bf b = g_loadB(bp + (size_t)(s1 - 1) * 32);
    const unsigned* ab = &ldsA[buf][0];
    v16bf a0 = lds_loadA(ab + (ln     ) * 16 + hi * 4);
    v16bf a1 = lds_loadA(ab + (ln + 16) * 16 + hi * 4);
    acc0 = __builtin_amdgcn_wmma_f32_16x16x32_bf16(false, a0, false, b,
                                                   (short)0, acc0, false, false);
    acc1 = __builtin_amdgcn_wmma_f32_16x16x32_bf16(false, a1, false, b,
                                                   (short)0, acc1, false, false);
  }

  // C/D layout: VGPR j -> M = j + 8*hi, N = lane&15
  float* dst = parts + (size_t)ks * BATCH * DIM
                     + (size_t)(rowblk * 32) * DIM + col0 + ln;
#pragma unroll
  for (int j = 0; j < 8; ++j) {
    const int m = j + hi * 8;
    dst[(size_t)m * DIM]        = acc0[j];
    dst[(size_t)(m + 16) * DIM] = acc1[j];
  }
}

// ---- reduce split-K partials (fixed order -> deterministic) ----
__global__ __launch_bounds__(256) void fg_reduce(const float* __restrict__ parts,
                                                 float* __restrict__ aggf) {
  const int i = blockIdx.x * 256 + threadIdx.x;
  if (i < BATCH * DIM) {
    float s = 0.f;
#pragma unroll
    for (int p = 0; p < KSPLIT; ++p) s += parts[(size_t)p * BATCH * DIM + i];
    aggf[i] = s;
  }
}

// ---- linear: out = agg @ W^T + bias, bf16 WMMA, fused bias epilogue ----
__global__ __launch_bounds__(256) void fg_linear(const float* __restrict__ aggf,
                                                 const __bf16* __restrict__ wbf,
                                                 const float* __restrict__ bias,
                                                 float* __restrict__ out) {
  const int tid = threadIdx.x;
  const int rowt = blockIdx.x;                // 0..127 (16 rows each)
  const int lane = tid & 31, wave = tid >> 5;
  const int ln = lane & 15, hi = lane >> 4;
  const int col0 = wave * 16;

  const float*  arow = aggf + (size_t)(rowt * 16 + ln) * DIM;
  const __bf16* bp   = wbf  + (size_t)(col0 + ln) * DIM + (size_t)hi * 8;

  v8f acc = {0.f,0.f,0.f,0.f,0.f,0.f,0.f,0.f};
#pragma unroll
  for (int kb = 0; kb < 4; ++kb) {
    const int kbase = kb * 32;
    float t[16];
    *(float4*)&t[0]  = *(const float4*)(arow + kbase + hi * 8);
    *(float4*)&t[4]  = *(const float4*)(arow + kbase + hi * 8 + 4);
    *(float4*)&t[8]  = *(const float4*)(arow + kbase + 16 + hi * 8);
    *(float4*)&t[12] = *(const float4*)(arow + kbase + 16 + hi * 8 + 4);
    v16bf a;
#pragma unroll
    for (int e = 0; e < 16; ++e) a[e] = (__bf16)t[e];
    v16bf b = g_loadB(bp + kbase);
    acc = __builtin_amdgcn_wmma_f32_16x16x32_bf16(false, a, false, b,
                                                  (short)0, acc, false, false);
  }
  const float bv = bias[col0 + ln];
  float* orow = out + (size_t)(rowt * 16) * DIM + col0 + ln;
#pragma unroll
  for (int j = 0; j < 8; ++j)
    orow[(size_t)(j + hi * 8) * DIM] = acc[j] + bv;
}

extern "C" void kernel_launch(void* const* d_in, const int* in_sizes, int n_in,
                              void* d_out, int out_size, void* d_ws, size_t ws_size,
                              hipStream_t stream) {
  (void)in_sizes; (void)n_in; (void)out_size; (void)ws_size;
  const int*   ids  = (const int*)  d_in[0];
  const float* F    = (const float*)d_in[1];
  const float* E    = (const float*)d_in[2];
  const float* W    = (const float*)d_in[3];
  const float* bias = (const float*)d_in[4];
  float* out = (float*)d_out;

  char* ws = (char*)d_ws;
  __bf16* ebf   = (__bf16*)ws;
  __bf16* wbf   = (__bf16*)(ws + EBF_BYTES);
  float*  parts = (float*) (ws + EBF_BYTES + WBF_BYTES);
  float*  aggf  = (float*) (ws + EBF_BYTES + WBF_BYTES + PARTS_BYTES);

  fg_prepE <<<dim3(KPAD / 16, DIM / 16), 256, 0, stream>>>(E, ebf);
  fg_prepW <<<(DIM * DIM + 255) / 256,   256, 0, stream>>>(W, wbf);
  fg_main  <<<dim3(BATCH / 32, KSPLIT),  256, 0, stream>>>(ids, F, ebf, parts);
  fg_reduce<<<(BATCH * DIM + 255) / 256, 256, 0, stream>>>(parts, aggf);
  fg_linear<<<BATCH / 16,                256, 0, stream>>>(aggf, wbf, bias, out);
}
